// NSMCell_7799660610214
// MI455X (gfx1250) — matrix operand order repaired and checked by reading, not compiled, not run on hardware
//
#include <hip/hip_runtime.h>
#include <math.h>

// ---------------------------------------------------------------------------
// NSM edge/relation branch (ins_id == 1) for gfx1250 / MI455X.
// Workgroup-cooperative bf16 WMMA GEMM:
//   block = 320 threads (10 waves); block owns one 16-row tile of
//   x = (instr * edge_attr) @ w_edge ; each wave owns exactly 2 of 20
//   (zero-padded) col tiles -> uniform, branch-free inner loop, 16 VGPR acc.
//   A fragment staged in LDS as bf16 (K padded 300->320, masking done at
//   staging); B operands from pre-transposed/padded bf16 WT (L2-resident).
//   k-loop software-pipelined by one step to overlap loads with WMMA.
// Two GEMM passes (column norms, then sigmoid/aggregate) avoid the 384MB
// intermediate. Deterministic: no float atomics, fixed-order reductions.
// ---------------------------------------------------------------------------

typedef __attribute__((ext_vector_type(16))) __bf16 v16bf;
typedef __attribute__((ext_vector_type(8)))  __bf16 v8bf;
typedef __attribute__((ext_vector_type(8)))  float  v8f;

#define Bc   8
#define Nn   200
#define Hh   300
#define NNr  40000      // Nn*Nn rows per batch
#define KP   320        // padded K (10 k-steps of 32)
#define CP   320        // padded output columns (20 tiles * 16)
#define NT   20
#define AP   328        // LDS A-row stride in bf16 elems (bank-conflict pad)
#define TPB  320        // 10 waves per block
#define P1_CHUNKS 250   // blocks per batch in pass1
#define P1_TILES  10    // row tiles per pass1 block

__device__ __forceinline__ v8f wmma_bf16(v16bf a, v16bf b, v8f c) {
  return __builtin_amdgcn_wmma_f32_16x16x32_bf16(false, a, false, b,
                                                 (short)0, c, false, false);
}

// Cooperative staging of one A tile: 16 rows x 320 K, bf16, scaled by instr.
// thread -> (row = tid/20, k-chunk = (tid%20)*16). K >= 300 and invalid rows
// are written as zero, so the MMA loop below needs no masking at all.
__device__ __forceinline__ void stageA(const float* __restrict__ ea_tile,
                                       const float* __restrict__ insb,
                                       __bf16* __restrict__ Alds,
                                       int tid, int validRows)
{
  const int r  = tid / 20;            // 0..15
  const int kc = (tid % 20) * 16;     // 0,16,...,304
  const bool rv = r < validRows;
  const float* rowp = ea_tile + (long)r * Hh;
  __bf16 tmp[16];
  if (rv && (kc + 16) <= Hh) {        // fast path: kc <= 272
    #pragma unroll
    for (int i = 0; i < 16; i += 4) {
      const float4 e = *(const float4*)(rowp + kc + i);
      const float4 s = *(const float4*)(insb + kc + i);
      tmp[i + 0] = (__bf16)(e.x * s.x);
      tmp[i + 1] = (__bf16)(e.y * s.y);
      tmp[i + 2] = (__bf16)(e.z * s.z);
      tmp[i + 3] = (__bf16)(e.w * s.w);
    }
  } else {                            // K tail (kc==288) / padding / masked rows
    #pragma unroll
    for (int i = 0; i < 16; ++i) {
      const int k = kc + i;
      const float v = (rv && k < Hh) ? rowp[k] * insb[k] : 0.f;
      tmp[i] = (__bf16)v;
    }
  }
  v8bf* dst = (v8bf*)(Alds + r * AP + kc);
  dst[0] = *(const v8bf*)(tmp);
  dst[1] = *(const v8bf*)(tmp + 8);
}

// Per-wave MMA over one staged A tile and this wave's 2 column tiles.
// A k order chosen here: lanes 0-15 hold K 0-15, lanes 16-31 K 16-31
// per 32-wide k-step; lane row M = lane&15. B: col = ct*16 + (lane&15),
// 16 consecutive K per lane (hi lanes +16) -> one 32B load from transposed WT.
// Software-pipelined by one k-step.
__device__ __forceinline__ void tileMMA(const __bf16* __restrict__ Alds,
                                        const __bf16* __restrict__ WT,
                                        int ln, int hi, int ct0,
                                        v8f& acc0, v8f& acc1)
{
  const __bf16* ap  = Alds + ln * AP + hi * 16;
  const __bf16* wt0 = WT + (ct0 * 16 + ln) * KP + hi * 16;
  const __bf16* wt1 = wt0 + 16 * KP;
  v8bf  alo = *(const v8bf*)(ap);
  v8bf  ahi = *(const v8bf*)(ap + 8);
  v16bf b0  = *(const v16bf*)(wt0);
  v16bf b1  = *(const v16bf*)(wt1);
  #pragma unroll 1
  for (int kk = 0; kk < 9; ++kk) {
    // issue next k-step's loads before consuming the current operands
    const v8bf  alo_n = *(const v8bf*)(ap + (kk + 1) * 32);
    const v8bf  ahi_n = *(const v8bf*)(ap + (kk + 1) * 32 + 8);
    const v16bf b0_n  = *(const v16bf*)(wt0 + (kk + 1) * 32);
    const v16bf b1_n  = *(const v16bf*)(wt1 + (kk + 1) * 32);
    const v16bf a = __builtin_shufflevector(alo, ahi,
        0,1,2,3,4,5,6,7,8,9,10,11,12,13,14,15);
    acc0 = wmma_bf16(a, b0, acc0);
    acc1 = wmma_bf16(a, b1, acc1);
    alo = alo_n; ahi = ahi_n; b0 = b0_n; b1 = b1_n;
  }
  const v16bf a = __builtin_shufflevector(alo, ahi,
      0,1,2,3,4,5,6,7,8,9,10,11,12,13,14,15);
  acc0 = wmma_bf16(a, b0, acc0);
  acc1 = wmma_bf16(a, b1, acc1);
}

// WT[j*KP + k] = bf16(w_edge[k*300 + j]), zero padded to 320 x 320
__global__ void initWT_kernel(const float* __restrict__ w_edge,
                              __bf16* __restrict__ WT)
{
  const int idx = blockIdx.x * blockDim.x + threadIdx.x;
  if (idx >= CP * KP) return;
  const int j = idx / KP, k = idx % KP;
  const float v = (j < Hh && k < Hh) ? w_edge[k * Hh + j] : 0.f;
  WT[idx] = (__bf16)v;
}

// Pass 1: per-(b,h) sum of squares over all 40000 rows -> per-chunk partials
__global__ __launch_bounds__(TPB)
void pass1_kernel(const float* __restrict__ ea,
                  const float* __restrict__ ins,
                  const __bf16* __restrict__ WT,
                  float* __restrict__ partials)
{
  __shared__ __bf16 Alds[16 * AP];
  const int tid = threadIdx.x;
  const int w = tid >> 5, lane = tid & 31;
  const int hi = lane >> 4, ln = lane & 15;
  const int b = blockIdx.x / P1_CHUNKS, chunk = blockIdx.x % P1_CHUNKS;
  const float* insb = ins + b * Hh;
  const int ct0 = 2 * w;

  float ssq0 = 0.f, ssq1 = 0.f;
  #pragma unroll 1
  for (int t = 0; t < P1_TILES; ++t) {
    const long tile = (long)chunk * P1_TILES + t;
    __syncthreads();                                  // protect previous reads
    stageA(ea + ((long)b * NNr + tile * 16) * Hh, insb, Alds, tid, 16);
    __syncthreads();
    v8f acc0 = {0.f,0.f,0.f,0.f,0.f,0.f,0.f,0.f};
    v8f acc1 = {0.f,0.f,0.f,0.f,0.f,0.f,0.f,0.f};
    tileMMA(Alds, WT, ln, hi, ct0, acc0, acc1);
    #pragma unroll
    for (int i = 0; i < 8; ++i) {
      ssq0 += acc0[i] * acc0[i];
      ssq1 += acc1[i] * acc1[i];
    }
  }
  const float v0 = ssq0 + __shfl_xor(ssq0, 16, 32);   // merge half-waves
  const float v1 = ssq1 + __shfl_xor(ssq1, 16, 32);
  if (hi == 0) {
    float* dst = partials + ((long)b * P1_CHUNKS + chunk) * CP;
    dst[ct0 * 16 + ln] = v0;
    dst[(ct0 + 1) * 16 + ln] = v1;
  }
}

__global__ void reduce_norms_kernel(const float* __restrict__ partials,
                                    float* __restrict__ norms)
{
  const int idx = blockIdx.x * blockDim.x + threadIdx.x;
  if (idx >= Bc * Hh) return;
  const int b = idx / Hh, j = idx % Hh;
  float s = 0.f;
  for (int c = 0; c < P1_CHUNKS; ++c)
    s += partials[((long)b * P1_CHUNKS + c) * CP + j];
  norms[idx] = s;                                     // sum of squares
}

// Pass 2: block = (b, n). Recompute GEMM for the 200 source rows of node n,
// apply sigmoid(x/||col||), weight by distribution[b,m], reduce over m.
__global__ __launch_bounds__(TPB)
void pass2_kernel(const float* __restrict__ ea,
                  const float* __restrict__ ins,
                  const float* __restrict__ dist,
                  const __bf16* __restrict__ WT,
                  const float* __restrict__ norms,
                  float* __restrict__ agg)
{
  __shared__ __bf16 Alds[16 * AP];
  const int tid = threadIdx.x;
  const int w = tid >> 5, lane = tid & 31;
  const int hi = lane >> 4, ln = lane & 15;
  const int b = blockIdx.x / Nn, n = blockIdx.x % Nn;
  const float* insb = ins + b * Hh;
  const float* db   = dist + b * Nn;
  const int ct0 = 2 * w;

  float rn0, rn1;
  {
    const int j0 = ct0 * 16 + ln, j1 = j0 + 16;
    rn0 = 1.f / fmaxf(sqrtf((j0 < Hh) ? norms[b * Hh + j0] : 1.f), 1e-12f);
    rn1 = 1.f / fmaxf(sqrtf((j1 < Hh) ? norms[b * Hh + j1] : 1.f), 1e-12f);
  }

  float ws0 = 0.f, ws1 = 0.f;
  #pragma unroll 1
  for (int mt = 0; mt < 13; ++mt) {                   // 13*16 = 208 >= 200 rows
    const int validRows = (Nn - mt * 16 < 16) ? (Nn - mt * 16) : 16;
    __syncthreads();
    stageA(ea + ((long)b * NNr + (long)n * Nn + mt * 16) * Hh,
           insb, Alds, tid, validRows);
    __syncthreads();
    v8f acc0 = {0.f,0.f,0.f,0.f,0.f,0.f,0.f,0.f};
    v8f acc1 = {0.f,0.f,0.f,0.f,0.f,0.f,0.f,0.f};
    tileMMA(Alds, WT, ln, hi, ct0, acc0, acc1);
    // C/D rows: lanes 0-15 hold M=i, lanes 16-31 hold M=i+8
    #pragma unroll
    for (int i = 0; i < 8; ++i) {
      const int m = mt * 16 + i + 8 * hi;
      const float g = (m < Nn) ? db[m] : 0.f;
      ws0 += g / (1.f + __expf(-acc0[i] * rn0));      // g * sigmoid(x0)
      ws1 += g / (1.f + __expf(-acc1[i] * rn1));
    }
  }
  const float v0 = ws0 + __shfl_xor(ws0, 16, 32);
  const float v1 = ws1 + __shfl_xor(ws1, 16, 32);
  if (hi == 0) {
    float* dst = agg + ((long)b * Nn + n) * CP;       // exclusive per block
    dst[ct0 * 16 + ln] = v0;
    dst[(ct0 + 1) * 16 + ln] = v1;
  }
}

// Pass 3: s = agg @ w_rel, softmax over n, one block per batch
__global__ void pass3_kernel(const float* __restrict__ agg,
                             const float* __restrict__ w_rel,
                             float* __restrict__ out)
{
  __shared__ float red[256];
  const int b = blockIdx.x, t = threadIdx.x;
  float sval = 0.f;
  if (t < Nn) {
    const float* a0 = agg + ((long)b * Nn + t) * CP;
    float d = 0.f;
    for (int j = 0; j < Hh; ++j) d += a0[j] * w_rel[j];
    sval = d;
  }
  red[t] = (t < Nn) ? sval : -3.4e38f;
  __syncthreads();
  for (int s = 128; s > 0; s >>= 1) {
    if (t < s) red[t] = fmaxf(red[t], red[t + s]);
    __syncthreads();
  }
  const float mx = red[0];
  __syncthreads();
  const float e = (t < Nn) ? __expf(sval - mx) : 0.f;
  red[t] = e;
  __syncthreads();
  for (int s = 128; s > 0; s >>= 1) {
    if (t < s) red[t] += red[t + s];
    __syncthreads();
  }
  if (t < Nn) out[b * Nn + t] = e / red[0];
}

extern "C" void kernel_launch(void* const* d_in, const int* in_sizes, int n_in,
                              void* d_out, int out_size, void* d_ws, size_t ws_size,
                              hipStream_t stream)
{
  // setup_inputs order: node_attr, edge_attr, instruction, distribution,
  //                     w_node_props, w_edge, w_state, w_rel, ins_id (== 1)
  const float* ea     = (const float*)d_in[1];
  const float* ins    = (const float*)d_in[2];
  const float* dist   = (const float*)d_in[3];
  const float* w_edge = (const float*)d_in[5];
  const float* w_rel  = (const float*)d_in[7];
  float* out = (float*)d_out;
  (void)in_sizes; (void)n_in; (void)out_size; (void)ws_size;

  // workspace layout (~4.9 MB)
  char* ws = (char*)d_ws;
  __bf16* WT    = (__bf16*)ws;                               // 320*320 bf16
  float*  part1 = (float*)(ws + (size_t)CP * KP * 2);        // 8*250*320 f32
  float*  norms = part1 + (size_t)Bc * P1_CHUNKS * CP;       // 2400 f32
  float*  agg   = norms + (size_t)Bc * Hh;                   // 8*200*320 f32

  initWT_kernel<<<(CP * KP + 255) / 256, 256, 0, stream>>>(w_edge, WT);
  // pass1: 8 batches * 250 chunks, 10 row-tiles per block
  pass1_kernel<<<Bc * P1_CHUNKS, TPB, 0, stream>>>(ea, ins, WT, part1);
  reduce_norms_kernel<<<(Bc * Hh + 255) / 256, 256, 0, stream>>>(part1, norms);
  // pass2: one block per (b, n)
  pass2_kernel<<<Bc * Nn, TPB, 0, stream>>>(ea, ins, dist, WT, norms, agg);
  pass3_kernel<<<Bc, 256, 0, stream>>>(agg, w_rel, out);
}
